// SageConv_53489522704385
// MI455X (gfx1250) — matrix operand order repaired and checked by reading, not compiled, not run on hardware
//
#include <hip/hip_runtime.h>
#include <hip/hip_bf16.h>

// ---------- CDNA5 WMMA types ----------
typedef __attribute__((ext_vector_type(16))) __bf16 v16bf;
typedef __attribute__((ext_vector_type(8)))  __bf16 bf16x8;
typedef __attribute__((ext_vector_type(4)))  __bf16 bf16x4;
typedef __attribute__((ext_vector_type(8)))  float  v8f;

#define WAVES_PER_BLOCK 8
#define BLOCK_THREADS   (WAVES_PER_BLOCK * 32)

#define SHUF16(lo, hi) \
    __builtin_shufflevector(lo, hi, 0, 1, 2, 3, 4, 5, 6, 7, 8, 9, 10, 11, 12, 13, 14, 15)

__device__ __forceinline__ v8f wmma_bf16(v16bf a, v16bf b, v8f c) {
    return __builtin_amdgcn_wmma_f32_16x16x32_bf16(false, a, false, b, (short)0, c,
                                                   false, false);
}

template <int ACT>
__device__ __forceinline__ float act_apply(float v) {
    if (ACT == 1) return tanhf(v);
    if (ACT == 2) return fmaxf(v, 0.0f);
    if (ACT == 3) return 1.0f / (1.0f + expf(-v));
    return v;
}

// ---- Async copy of a packed B panel (global -> LDS), CDNA5 async-DMA path ----
// Each thread copies 16B units; tracked on ASYNCcnt, drained with s_wait_asynccnt.
template <int N128>
__device__ __forceinline__ void stage_panel(__bf16* sdst, const __bf16* __restrict__ gsrc,
                                            int tid) {
    unsigned lbase = (unsigned)(size_t)sdst;  // low 32 bits of generic ptr = LDS offset
#pragma unroll
    for (int u = 0; u < N128; u += BLOCK_THREADS) {
        unsigned boff = (unsigned)(u + tid) * 16u;
        asm volatile("global_load_async_to_lds_b128 %0, %1, %2"
                     :
                     : "v"(lbase + boff), "v"(boff), "s"(gsrc)
                     : "memory");
    }
}

__device__ __forceinline__ void async_drain_and_barrier() {
    asm volatile("s_wait_asynccnt 0x0" ::: "memory");
    __syncthreads();
}

// A-operand (16x32 bf16): lane L -> M = L&15, kbase = (L>=16)?8:0.
// VGPRs 0..3 hold K = kbase..kbase+7 (16 contiguous bytes), VGPRs 4..7 hold K = 16+kbase..+7.
__device__ __forceinline__ v16bf load_a_bf16(const __bf16* __restrict__ A, unsigned base,
                                             int half) {
    bf16x8 lo = *(const bf16x8*)(A + base + half * 8);
    bf16x8 hi = *(const bf16x8*)(A + base + 16 + half * 8);
    return SHUF16(lo, hi);
}

__device__ __forceinline__ v16bf load_a_f32(const float* __restrict__ A, unsigned base,
                                            int half, float s) {
    float4 f0 = *(const float4*)(A + base + half * 8);
    float4 f1 = *(const float4*)(A + base + half * 8 + 4);
    float4 f2 = *(const float4*)(A + base + 16 + half * 8);
    float4 f3 = *(const float4*)(A + base + 16 + half * 8 + 4);
    v16bf a;
    a[0] = (__bf16)(f0.x * s);  a[1] = (__bf16)(f0.y * s);
    a[2] = (__bf16)(f0.z * s);  a[3] = (__bf16)(f0.w * s);
    a[4] = (__bf16)(f1.x * s);  a[5] = (__bf16)(f1.y * s);
    a[6] = (__bf16)(f1.z * s);  a[7] = (__bf16)(f1.w * s);
    a[8]  = (__bf16)(f2.x * s); a[9]  = (__bf16)(f2.y * s);
    a[10] = (__bf16)(f2.z * s); a[11] = (__bf16)(f2.w * s);
    a[12] = (__bf16)(f3.x * s); a[13] = (__bf16)(f3.y * s);
    a[14] = (__bf16)(f3.z * s); a[15] = (__bf16)(f3.w * s);
    return a;
}

// B-operand from LDS panel: layout [nTileLocal][kb][lane][16 bf16] -> 2x ds_load_b128
template <int KTILES>
__device__ __forceinline__ v16bf load_b_lds(const __bf16* sB, int j, int kb, int lane) {
    const __bf16* p = sB + ((unsigned)((j * KTILES + kb) * 32 + lane)) * 16u;
    bf16x8 b0 = *(const bf16x8*)p;
    bf16x8 b1 = *(const bf16x8*)(p + 8);
    return SHUF16(b0, b1);
}

// ---------- Weight packing: f32 row-major [K,H] -> WMMA B-operand lane layout ----------
// B (32x16 bf16) lane layout: lane L -> N = L&15, kbase = (L>=16)?8:0; VGPR v -> K=kbase+2v(+8 if v>=4),+1
__global__ void pack_w(const float* __restrict__ W, __bf16* __restrict__ pack, int K,
                       int H) {
    int t = blockIdx.x * blockDim.x + threadIdx.x;
    int kTiles = K >> 5;
    int total = (H >> 4) * kTiles * 32;
    if (t >= total) return;
    int lane  = t & 31;
    int kb    = (t >> 5) % kTiles;
    int nTile = (t >> 5) / kTiles;
    int colB  = nTile * 16 + (lane & 15);
    int kbase = kb * 32 + ((lane >> 4) << 3);
    __bf16* dst = pack + (size_t)t * 16;
#pragma unroll
    for (int v = 0; v < 8; ++v) {
        int k = kbase + 2 * v + (v >= 4 ? 8 : 0);
        dst[2 * v]     = (__bf16)W[k * H + colB];
        dst[2 * v + 1] = (__bf16)W[(k + 1) * H + colB];
    }
}

// ---------- Layer 1: out = tanh( (agg*invdeg) @ Wl + x @ Wr + b ), bf16 out ----------
// Block = 8 m-tiles x 1 n-group (NT tiles). Both W panels staged in LDS via async copy.
template <int K, int H, int NT>
__global__ __launch_bounds__(BLOCK_THREADS) void gemm_layer1(
    const float* __restrict__ agg, const float* __restrict__ x,
    const float* __restrict__ invdeg, const __bf16* __restrict__ pWl,
    const __bf16* __restrict__ pWr, const float* __restrict__ bias,
    __bf16* __restrict__ out, int M) {
    constexpr int kTiles = K / 32;
    constexpr int ntg    = (H / 16) / NT;
    constexpr int PANEL  = NT * kTiles * 512;  // bf16 elements per panel
    __shared__ __bf16 sBl[PANEL];
    __shared__ __bf16 sBr[PANEL];

    int tid  = threadIdx.x;
    int lane = tid & 31;
    int wave = tid >> 5;
    int ng     = blockIdx.x % ntg;
    int mBlock = blockIdx.x / ntg;

    stage_panel<PANEL / 8>(sBl, pWl + (size_t)ng * PANEL, tid);
    stage_panel<PANEL / 8>(sBr, pWr + (size_t)ng * PANEL, tid);
    async_drain_and_barrier();

    unsigned m0 = (unsigned)(mBlock * WAVES_PER_BLOCK + wave) * 16u;
    if (m0 >= (unsigned)M) return;  // wave-uniform (after barrier)
    int half = lane >> 4;
    int lm   = lane & 15;
    unsigned row    = m0 + lm;
    unsigned rowOff = row * (unsigned)K;
    float rs = invdeg[row];

    v8f zero = {};
    v8f c[NT];
#pragma unroll
    for (int j = 0; j < NT; ++j) c[j] = zero;

#pragma unroll
    for (int kb = 0; kb < kTiles; ++kb) {
        v16bf a = load_a_f32(agg, rowOff + kb * 32, half, rs);
        v16bf b[NT];
#pragma unroll
        for (int j = 0; j < NT; ++j) b[j] = load_b_lds<kTiles>(sBl, j, kb, lane);
#pragma unroll
        for (int j = 0; j < NT; ++j) c[j] = wmma_bf16(a, b[j], c[j]);
    }
#pragma unroll
    for (int kb = 0; kb < kTiles; ++kb) {
        v16bf a = load_a_f32(x, rowOff + kb * 32, half, 1.0f);
        v16bf b[NT];
#pragma unroll
        for (int j = 0; j < NT; ++j) b[j] = load_b_lds<kTiles>(sBr, j, kb, lane);
#pragma unroll
        for (int j = 0; j < NT; ++j) c[j] = wmma_bf16(a, b[j], c[j]);
    }
#pragma unroll
    for (int j = 0; j < NT; ++j) {
        unsigned colB = (unsigned)(ng * NT + j) * 16u + lm;
        float bb = bias[colB];
#pragma unroll
        for (int v = 0; v < 8; ++v) {
            unsigned m = m0 + v + half * 8;
            out[m * (unsigned)H + colB] = (__bf16)act_apply<1>(c[j][v] + bb);
        }
    }
}

// ---------- Generic bf16-A GEMM: out = ACT( A @ W [+ addend*invdeg + bias] ) ----------
template <int K, int H, int NT, int ACT, bool OUT_F32, bool FUSE>
__global__ __launch_bounds__(BLOCK_THREADS) void gemm_bf16a(
    const __bf16* __restrict__ A, const __bf16* __restrict__ pW,
    const float* __restrict__ addend, const float* __restrict__ invdeg,
    const float* __restrict__ bias, void* __restrict__ outv, int M) {
    constexpr int kTiles = K / 32;
    constexpr int ntg    = (H / 16) / NT;
    constexpr int PANEL  = NT * kTiles * 512;
    __shared__ __bf16 sB[PANEL];

    int tid  = threadIdx.x;
    int lane = tid & 31;
    int wave = tid >> 5;
    int ng     = blockIdx.x % ntg;
    int mBlock = blockIdx.x / ntg;

    stage_panel<PANEL / 8>(sB, pW + (size_t)ng * PANEL, tid);
    async_drain_and_barrier();

    unsigned m0 = (unsigned)(mBlock * WAVES_PER_BLOCK + wave) * 16u;
    if (m0 >= (unsigned)M) return;  // wave-uniform (after barrier)
    int half = lane >> 4;
    int lm   = lane & 15;
    unsigned rowOff = (m0 + lm) * (unsigned)K;

    v8f zero = {};
    v8f c[NT];
#pragma unroll
    for (int j = 0; j < NT; ++j) c[j] = zero;

#pragma unroll
    for (int kb = 0; kb < kTiles; ++kb) {
        __builtin_prefetch(A + rowOff + kb * 32 + 32, 0, 1);
        v16bf a = load_a_bf16(A, rowOff + kb * 32, half);
        v16bf b[NT];
#pragma unroll
        for (int j = 0; j < NT; ++j) b[j] = load_b_lds<kTiles>(sB, j, kb, lane);
#pragma unroll
        for (int j = 0; j < NT; ++j) c[j] = wmma_bf16(a, b[j], c[j]);
    }
#pragma unroll
    for (int j = 0; j < NT; ++j) {
        unsigned colB = (unsigned)(ng * NT + j) * 16u + lm;
#pragma unroll
        for (int v = 0; v < 8; ++v) {
            unsigned m = m0 + v + half * 8;
            float val = c[j][v];
            if (FUSE) val += addend[m * (unsigned)H + colB] * invdeg[m] + bias[colB];
            val = act_apply<ACT>(val);
            if (OUT_F32)
                ((float*)outv)[m * (unsigned)H + colB] = val;
            else
                ((__bf16*)outv)[m * (unsigned)H + colB] = (__bf16)val;
        }
    }
}

// ---------- Graph aggregation ----------
__global__ void deg_kernel(const long long* __restrict__ dst, int E,
                           int* __restrict__ deg) {
    int i = blockIdx.x * blockDim.x + threadIdx.x;
    if (i < E) atomicAdd(&deg[(int)dst[i]], 1);
}

__global__ void invdeg_kernel(const int* __restrict__ degi, float* __restrict__ outf,
                              int N) {
    int i = blockIdx.x * blockDim.x + threadIdx.x;
    if (i < N) {
        int c = degi[i];
        outf[i] = 1.0f / (float)(c > 0 ? c : 1);
    }
}

// agg[dst[e]*D + 4f..] += feat[src[e]*D + 4f..]  (f32 source, float4 per thread)
template <int D>
__global__ void scatter_add_f32(const float* __restrict__ feat,
                                const long long* __restrict__ src,
                                const long long* __restrict__ dst,
                                float* __restrict__ agg, unsigned nwork) {
    constexpr unsigned D4 = D / 4;
    unsigned i = blockIdx.x * blockDim.x + threadIdx.x;
    if (i >= nwork) return;
    unsigned e  = i / D4;  // division by constant
    unsigned f4 = i - e * D4;
    unsigned s = (unsigned)src[e], t = (unsigned)dst[e];
    float4 v = *((const float4*)(feat + (size_t)s * D) + f4);
    float* q = agg + (size_t)t * D + f4 * 4;
    atomicAdd(q + 0, v.x);
    atomicAdd(q + 1, v.y);
    atomicAdd(q + 2, v.z);
    atomicAdd(q + 3, v.w);
}

// bf16 source variant: 8B vector load, f32 atomic accumulation
template <int D>
__global__ void scatter_add_bf16(const __bf16* __restrict__ feat,
                                 const long long* __restrict__ src,
                                 const long long* __restrict__ dst,
                                 float* __restrict__ agg, unsigned nwork) {
    constexpr unsigned D4 = D / 4;
    unsigned i = blockIdx.x * blockDim.x + threadIdx.x;
    if (i >= nwork) return;
    unsigned e  = i / D4;
    unsigned f4 = i - e * D4;
    unsigned s = (unsigned)src[e], t = (unsigned)dst[e];
    bf16x4 v = *((const bf16x4*)(feat + (size_t)s * D) + f4);
    float* q = agg + (size_t)t * D + f4 * 4;
    atomicAdd(q + 0, (float)v[0]);
    atomicAdd(q + 1, (float)v[1]);
    atomicAdd(q + 2, (float)v[2]);
    atomicAdd(q + 3, (float)v[3]);
}

static inline int iceil(long long a, long long b) { return (int)((a + b - 1) / b); }

extern "C" void kernel_launch(void* const* d_in, const int* in_sizes, int n_in,
                              void* d_out, int out_size, void* d_ws, size_t ws_size,
                              hipStream_t stream) {
    const float*     x    = (const float*)d_in[0];
    const long long* eidx = (const long long*)d_in[1];
    const float* W1l = (const float*)d_in[2];
    const float* W1r = (const float*)d_in[3];
    const float* b1  = (const float*)d_in[4];
    const float* W2l = (const float*)d_in[5];
    const float* W2r = (const float*)d_in[6];
    const float* b2  = (const float*)d_in[7];
    const float* W3l = (const float*)d_in[8];
    const float* W3r = (const float*)d_in[9];
    const float* b3  = (const float*)d_in[10];

    const int F_IN = 128, H1 = 256, H2 = 192, C = 32;
    int N = in_sizes[0] / F_IN;
    int E = in_sizes[1] / 2;
    const long long* srcI = eidx;
    const long long* dstI = eidx + E;

    // Workspace carve-out (~167 MB + ~350 KB packed weights)
    char* ws = (char*)d_ws;
    size_t off = 0;
    float* invdeg = (float*)(ws + off);  off += (size_t)N * 4;       // deg(int) then 1/deg(f32)
    float* agg    = (float*)(ws + off);  off += (size_t)N * H1 * 4;  // f32 scatter accumulators
    __bf16* h1    = (__bf16*)(ws + off); off += (size_t)N * H1 * 2;  // later reused as Q (N x 32)
    __bf16* h2    = (__bf16*)(ws + off); off += (size_t)N * H2 * 2;  // first holds P, then h2
    __bf16* pW1l  = (__bf16*)(ws + off); off += (size_t)F_IN * H1 * 2;
    __bf16* pW1r  = (__bf16*)(ws + off); off += (size_t)F_IN * H1 * 2;
    __bf16* pW2l  = (__bf16*)(ws + off); off += (size_t)H1 * H2 * 2;
    __bf16* pW2r  = (__bf16*)(ws + off); off += (size_t)H1 * H2 * 2;
    __bf16* pW3l  = (__bf16*)(ws + off); off += (size_t)H2 * C * 2;
    __bf16* pW3r  = (__bf16*)(ws + off); off += (size_t)H2 * C * 2;
    (void)off; (void)ws_size; (void)n_in; (void)out_size;

    // ---- pack weights into WMMA B-operand layout (tiny, L2-resident) ----
    {
        int t1 = (H1 / 16) * (F_IN / 32) * 32;
        pack_w<<<iceil(t1, 256), 256, 0, stream>>>(W1l, pW1l, F_IN, H1);
        pack_w<<<iceil(t1, 256), 256, 0, stream>>>(W1r, pW1r, F_IN, H1);
        int t2 = (H2 / 16) * (H1 / 32) * 32;
        pack_w<<<iceil(t2, 256), 256, 0, stream>>>(W2l, pW2l, H1, H2);
        pack_w<<<iceil(t2, 256), 256, 0, stream>>>(W2r, pW2r, H1, H2);
        int t3 = (C / 16) * (H2 / 32) * 32;
        pack_w<<<iceil(t3, 256), 256, 0, stream>>>(W3l, pW3l, H2, C);
        pack_w<<<iceil(t3, 256), 256, 0, stream>>>(W3r, pW3r, H2, C);
    }

    // ---- degree / inverse degree (shared by all layers) ----
    hipMemsetAsync(invdeg, 0, (size_t)N * 4, stream);
    deg_kernel<<<iceil(E, 256), 256, 0, stream>>>(dstI, E, (int*)invdeg);
    invdeg_kernel<<<iceil(N, 256), 256, 0, stream>>>((const int*)invdeg, invdeg, N);

    int mBlocks = iceil((N + 15) / 16, WAVES_PER_BLOCK);

    // ---- Layer 1: aggregate x (d=128); fused dual-GEMM + tanh -> h1 (bf16) ----
    hipMemsetAsync(agg, 0, (size_t)N * F_IN * 4, stream);
    {
        unsigned nwork = (unsigned)E * (F_IN / 4);
        scatter_add_f32<F_IN><<<iceil(nwork, 256), 256, 0, stream>>>(x, srcI, dstI, agg,
                                                                     nwork);
    }
    gemm_layer1<F_IN, H1, 4><<<mBlocks * ((H1 / 16) / 4), BLOCK_THREADS, 0, stream>>>(
        agg, x, invdeg, pW1l, pW1r, b1, h1, N);

    // ---- Layer 2: P = h1 @ W2_l; aggregate P (d=192); h2 = relu(h1@W2_r + mean + b2) ----
    __bf16* P = h2;  // P occupies the h2 buffer; consumed before h2 is written
    gemm_bf16a<H1, H2, 4, 0, false, false>
        <<<mBlocks * ((H2 / 16) / 4), BLOCK_THREADS, 0, stream>>>(
            h1, pW2l, nullptr, nullptr, nullptr, P, N);
    hipMemsetAsync(agg, 0, (size_t)N * H2 * 4, stream);
    {
        unsigned nwork = (unsigned)E * (H2 / 4);
        scatter_add_bf16<H2><<<iceil(nwork, 256), 256, 0, stream>>>(P, srcI, dstI, agg,
                                                                    nwork);
    }
    gemm_bf16a<H1, H2, 4, 2, false, true>
        <<<mBlocks * ((H2 / 16) / 4), BLOCK_THREADS, 0, stream>>>(
            h1, pW2r, agg, invdeg, b2, h2, N);

    // ---- Layer 3: Q = h2 @ W3_l; aggregate Q (d=32); out = sigmoid(h2@W3_r + mean + b3) ----
    __bf16* Q = h1;  // h1 dead after layer 2
    gemm_bf16a<H2, C, 2, 0, false, false>
        <<<mBlocks * ((C / 16) / 2), BLOCK_THREADS, 0, stream>>>(
            h2, pW3l, nullptr, nullptr, nullptr, Q, N);
    hipMemsetAsync(agg, 0, (size_t)N * C * 4, stream);
    {
        unsigned nwork = (unsigned)E * (C / 4);
        scatter_add_bf16<C><<<iceil(nwork, 256), 256, 0, stream>>>(Q, srcI, dstI, agg,
                                                                   nwork);
    }
    gemm_bf16a<H2, C, 2, 3, true, true>
        <<<mBlocks * ((C / 16) / 2), BLOCK_THREADS, 0, stream>>>(
            h2, pW3r, agg, invdeg, b3, d_out, N);
}
